// BMNet_68891275428418
// MI455X (gfx1250) — matrix operand order repaired and checked by reading, not compiled, not run on hardware
//
#include <hip/hip_runtime.h>
#include <hip/hip_bf16.h>

#define LRELU 0.1f

typedef __attribute__((ext_vector_type(16))) _Float16 v16h;
typedef __attribute__((ext_vector_type(8)))  _Float16 v8h;
typedef __attribute__((ext_vector_type(8)))  float    v8f;

static inline int cdiv(int a, int b) { return (a + b - 1) / b; }

// ---------------------------------------------------------------------------
// Implicit-GEMM 3x3 conv via WMMA f32 = f16 x f16 + f32.
//   M = N*Ho*Wo (spatial), N' = Cout, K = Cin*9.
//   128 threads = 4 waves; tile = 32 spatial x 64 cout; K stepped by 64.
//   Each wave owns a 16-cout slice and computes two 16x16 C tiles with
//   4 independent accumulators (4 static v_wmma sites, pipelineable).
//   LDS double-buffered: stage chunk t+1 while WMMAs consume chunk t,
//   one barrier per iteration. All LDS traffic is 16B ds_load/store_b128
//   (72-half row pitch -> conflict-free). im2col index math is incremental
//   (no divisions in the hot loop); bounds handling is fully branchless:
//   loads are unconditional from clamped coords, values zeroed by cndmask.
// ---------------------------------------------------------------------------
__global__ void __launch_bounds__(128)
conv3x3_wmma(const float* __restrict__ x, const float* __restrict__ w,
             const float* __restrict__ bias, float* __restrict__ out,
             int N, int Cin, int CinT, int ciOff, int H, int W,
             int Cout, int CoutT, int coOff, int Ho, int Wo,
             int stride, int pad, int dil, int act)
{
    __shared__ alignas(16) _Float16 Asm[2][32][72];   // M x K chunk (x2 buffers)
    __shared__ alignas(16) _Float16 Bsm[2][64][72];   // N x K chunk (transposed)

    const int tid   = threadIdx.x;
    const int wave  = tid >> 5;
    const int lane  = tid & 31;
    const int sBase = blockIdx.x * 32;
    const int nBase = blockIdx.y * 64;
    const int coW   = nBase + wave * 16;
    const int HWo   = Ho * Wo;
    const int totalS = N * HWo;
    const int K = Cin * 9;

    // ---- hoisted staging geometry: A covers 32 rows x 8 vec8 groups ----
    int am[2], ak[2], aoh[2], aow[2], aOK[2];
    const float* axp[2];
    int aci[2], akh[2], akw[2];          // incremental (ci, kh, kw) per group
    #pragma unroll
    for (int i = 0; i < 2; ++i) {
        int g = tid + i * 128;           // 0..255
        am[i] = g >> 3;                  // 0..31
        ak[i] = (g & 7) * 8;             // 0,8,..,56
        int s = sBase + am[i];
        aOK[i] = (s < totalS);
        int sc = aOK[i] ? s : (totalS - 1);   // clamp, keep math unconditional
        int n = sc / HWo;
        int rem = sc - n * HWo;
        aoh[i] = rem / Wo;
        aow[i] = rem - aoh[i] * Wo;
        axp[i] = x + ((size_t)n * CinT + ciOff) * (size_t)H * W;
        int ci = ak[i] / 9;
        int tap = ak[i] - ci * 9;
        aci[i] = ci; akh[i] = tap / 3; akw[i] = tap - (tap / 3) * 3;
    }
    // ---- B covers 64 rows x 8 vec8 groups ----
    int bn[4], bk[4], bOK[4];
    const float* wrp[4];
    #pragma unroll
    for (int i = 0; i < 4; ++i) {
        int g = tid + i * 128;           // 0..511
        bn[i] = g >> 3;                  // 0..63
        bk[i] = (g & 7) * 8;
        int co = nBase + bn[i];
        bOK[i] = (co < Cout);
        wrp[i] = w + (size_t)(bOK[i] ? co : 0) * K;
    }

    v8f acc[2][2] = {};   // [mtile][ksub]

    // branchless im2col gather; (ci,kh,kw) advanced incrementally;
    // loads unconditional from clamped coords, value zeroed via select
    auto stageA = [&](int k0, int buf) {
        #pragma unroll
        for (int i = 0; i < 2; ++i) {
            int ci = aci[i], kh = akh[i], kw = akw[i];
            v8h hv;
            #pragma unroll
            for (int j = 0; j < 8; ++j) {
                int ih = aoh[i] * stride - pad + kh * dil;
                int iw = aow[i] * stride - pad + kw * dil;
                int kg = k0 + ak[i] + j;
                int ok = aOK[i] & (ih >= 0) & (ih < H) & (iw >= 0) & (iw < W)
                       & (kg < K);
                int ihc = min(max(ih, 0), H - 1);
                int iwc = min(max(iw, 0), W - 1);
                int cic = min(ci, Cin - 1);
                float t = axp[i][((size_t)cic * H + ihc) * W + iwc];
                hv[j] = (_Float16)(ok ? t : 0.f);
                if (++kw == 3) { kw = 0; if (++kh == 3) { kh = 0; ++ci; } }
            }
            *(v8h*)&Asm[buf][am[i]][ak[i]] = hv;
            // advance remaining 56 taps to next chunk: 56 = 6*9 + 2
            ci += 6; kw += 2;
            if (kw >= 3) { kw -= 3; ++kh; }
            if (kh >= 3) { kh -= 3; ++ci; }
            aci[i] = ci; akh[i] = kh; akw[i] = kw;
        }
    };
    auto stageB = [&](int k0, int buf) {
        #pragma unroll
        for (int i = 0; i < 4; ++i) {
            int kg0 = k0 + bk[i];
            v8h hv;
            #pragma unroll
            for (int j = 0; j < 8; ++j) {
                int kg = kg0 + j;
                int ok = bOK[i] & (kg < K);
                int kc = min(kg, K - 1);
                float t = wrp[i][kc];
                hv[j] = (_Float16)(ok ? t : 0.f);
            }
            *(v8h*)&Bsm[buf][bn[i]][bk[i]] = hv;
        }
        if (k0 + 64 < K)
            __builtin_prefetch(&wrp[0][k0 + 64], 0, 0);
    };
    auto compute = [&](int buf) {
        const int ml = lane & 15;
        const int hi = lane >> 4;
        const int nn = wave * 16 + ml;
        #pragma unroll
        for (int sub = 0; sub < 2; ++sub) {
            const int kb = sub * 32;
            v8h b0 = *(const v8h*)&Bsm[buf][nn][kb + hi * 8];
            v8h b1 = *(const v8h*)&Bsm[buf][nn][kb + 16 + hi * 8];
            v16h bf = __builtin_shufflevector(b0, b1, 0, 1, 2, 3, 4, 5, 6, 7,
                                              8, 9, 10, 11, 12, 13, 14, 15);
            #pragma unroll
            for (int mt = 0; mt < 2; ++mt) {
                v8h a0 = *(const v8h*)&Asm[buf][mt * 16 + ml][kb + hi * 8];
                v8h a1 = *(const v8h*)&Asm[buf][mt * 16 + ml][kb + 16 + hi * 8];
                v16h af = __builtin_shufflevector(a0, a1, 0, 1, 2, 3, 4, 5, 6, 7,
                                                  8, 9, 10, 11, 12, 13, 14, 15);
                acc[mt][sub] = __builtin_amdgcn_wmma_f32_16x16x32_f16(
                    false, af, false, bf, (short)0, acc[mt][sub], false, false);
            }
        }
    };

    // ---- software-pipelined K loop: stage t+1 while computing t ----
    const int T = (K + 63) >> 6;
    stageA(0, 0);
    stageB(0, 0);
    __syncthreads();
    for (int t = 0; t < T; ++t) {
        const int cb = t & 1;
        if (t + 1 < T) {
            stageA((t + 1) * 64, cb ^ 1);
            stageB((t + 1) * 64, cb ^ 1);
        }
        compute(cb);
        __syncthreads();
    }

    // ---- epilogue: C layout lane0-15 -> M=r, lane16-31 -> M=8+r ----
    const int co = coW + (lane & 15);
    if (co < Cout) {
        const float bv = bias[co];
        const int mb = (lane >> 4) * 8;
        #pragma unroll
        for (int mt = 0; mt < 2; ++mt) {
            #pragma unroll
            for (int r = 0; r < 8; ++r) {
                int s = sBase + mt * 16 + mb + r;
                if (s < totalS) {
                    float v = acc[mt][0][r] + acc[mt][1][r] + bv;
                    if (act) v = (v > 0.f) ? v : LRELU * v;
                    int n   = s / HWo;
                    int rem = s - n * HWo;
                    int oh = rem / Wo, ow = rem - oh * Wo;
                    out[(((size_t)n * CoutT + coOff + co) * Ho + oh) * Wo + ow] = v;
                }
            }
        }
    }
}

// ---------------------------------------------------------------------------
// Channel-wise L2 normalize (x / sqrt(sum_c x^2 + 1e-6)), one thread per pixel
// ---------------------------------------------------------------------------
__global__ void l2norm_k(const float* __restrict__ x, float* __restrict__ y,
                         int N, int C, int H, int W)
{
    int i = blockIdx.x * blockDim.x + threadIdx.x;
    int HW = H * W, tot = N * HW;
    if (i >= tot) return;
    int n = i / HW, p = i - n * HW;
    const float* xb = x + (size_t)n * C * HW + p;
    float s = 1e-6f;
    for (int c = 0; c < C; ++c) { float v = xb[(size_t)c * HW]; s += v * v; }
    float r = rsqrtf(s);
    float* yb = y + (size_t)n * C * HW + p;
    for (int c = 0; c < C; ++c) yb[(size_t)c * HW] = xb[(size_t)c * HW] * r;
}

// align_corners=False normalize/unnormalize round-trip + border clamp + mask
__device__ inline void warp_setup(float fx, float fy, int ox, int oy, int H, int W,
                                  int& x0c, int& y0c, int& x1c, int& y1c,
                                  float& w00, float& w01, float& w10, float& w11,
                                  float& mask)
{
    float vx = (float)ox + fx, vy = (float)oy + fy;
    float nx = 2.f * vx / fmaxf((float)(W - 1), 1.f) - 1.f;
    float ny = 2.f * vy / fmaxf((float)(H - 1), 1.f) - 1.f;
    float px = ((nx + 1.f) * (float)W - 1.f) * 0.5f;
    float py = ((ny + 1.f) * (float)H - 1.f) * 0.5f;
    float x0f = floorf(px), y0f = floorf(py);
    float wx = px - x0f, wy = py - y0f;
    int x0 = (int)x0f, y0 = (int)y0f, x1 = x0 + 1, y1 = y0 + 1;
    w00 = (1.f - wx) * (1.f - wy); w01 = wx * (1.f - wy);
    w10 = (1.f - wx) * wy;         w11 = wx * wy;
    float i00 = (x0 >= 0 && x0 < W && y0 >= 0 && y0 < H) ? 1.f : 0.f;
    float i01 = (x1 >= 0 && x1 < W && y0 >= 0 && y0 < H) ? 1.f : 0.f;
    float i10 = (x0 >= 0 && x0 < W && y1 >= 0 && y1 < H) ? 1.f : 0.f;
    float i11 = (x1 >= 0 && x1 < W && y1 >= 0 && y1 < H) ? 1.f : 0.f;
    float m = w00 * i00 + w01 * i01 + w10 * i10 + w11 * i11;
    mask = (m < 0.999f) ? 0.f : 1.f;
    x0c = min(max(x0, 0), W - 1); x1c = min(max(x1, 0), W - 1);
    y0c = min(max(y0, 0), H - 1); y1c = min(max(y1, 0), H - 1);
}

// warp all C channels of x by fscale*bm, write into channel slice of out
__global__ void warp_k(const float* __restrict__ x, const float* __restrict__ bm,
                       float* __restrict__ out, int N, int C, int H, int W,
                       float fscale, int CoutT, int coOff)
{
    int i = blockIdx.x * blockDim.x + threadIdx.x;
    int HW = H * W, tot = N * HW;
    if (i >= tot) return;
    int n = i / HW, p = i - n * HW;
    int oy = p / W, ox = p - oy * W;
    float fx = fscale * bm[((size_t)n * 2 + 0) * HW + p];
    float fy = fscale * bm[((size_t)n * 2 + 1) * HW + p];
    int x0, y0, x1, y1; float w00, w01, w10, w11, mask;
    warp_setup(fx, fy, ox, oy, H, W, x0, y0, x1, y1, w00, w01, w10, w11, mask);
    const float* xb = x + (size_t)n * C * HW;
    float* ob = out + ((size_t)n * CoutT + coOff) * HW + p;
    for (int c = 0; c < C; ++c) {
        const float* pc = xb + (size_t)c * HW;
        float v = w00 * pc[y0 * W + x0] + w01 * pc[y0 * W + x1]
                + w10 * pc[y1 * W + x0] + w11 * pc[y1 * W + x1];
        ob[(size_t)c * HW] = v * mask;
    }
}

// Bilateral cost volume: one thread per (n, r, pixel); f1n/f2n pre-normalized.
// bw = warp(f1n, -2t*(bm+d)), fw = warp(f2n, +2(1-t)*(bm+d)); t = 0.5.
__global__ void costvol_k(const float* __restrict__ f1n, const float* __restrict__ f2n,
                          const float* __restrict__ bm, float* __restrict__ out,
                          int N, int C, int H, int W, int md,
                          int CoutT, int coOff, int act)
{
    int K = 2 * md + 1, R = K * K;
    int HW = H * W;
    int i = blockIdx.x * blockDim.x + threadIdx.x;
    int tot = N * R * HW;
    if (i >= tot) return;
    int p = i % HW; int t = i / HW; int r = t % R; int n = t / R;
    int oy = p / W, ox = p - oy * W;
    float bmx = 0.f, bmy = 0.f;
    if (bm) { bmx = bm[((size_t)n * 2 + 0) * HW + p]; bmy = bm[((size_t)n * 2 + 1) * HW + p]; }
    float mx = bmx + (float)(r % K - md);
    float my = bmy + (float)(r / K - md);
    int bx0, by0, bx1, by1; float bw00, bw01, bw10, bw11, bmask;
    int fx0, fy0, fx1, fy1; float fw00, fw01, fw10, fw11, fmask;
    warp_setup(-mx, -my, ox, oy, H, W, bx0, by0, bx1, by1, bw00, bw01, bw10, bw11, bmask);
    warp_setup( mx,  my, ox, oy, H, W, fx0, fy0, fx1, fy1, fw00, fw01, fw10, fw11, fmask);
    const float* f1b = f1n + (size_t)n * C * HW;
    const float* f2b = f2n + (size_t)n * C * HW;
    float s = 0.f;
    for (int c = 0; c < C; ++c) {
        const float* p1 = f1b + (size_t)c * HW;
        const float* p2 = f2b + (size_t)c * HW;
        float a = bw00 * p1[by0 * W + bx0] + bw01 * p1[by0 * W + bx1]
                + bw10 * p1[by1 * W + bx0] + bw11 * p1[by1 * W + bx1];
        float b = fw00 * p2[fy0 * W + fx0] + fw01 * p2[fy0 * W + fx1]
                + fw10 * p2[fy1 * W + fx0] + fw11 * p2[fy1 * W + fx1];
        s += a * b;
    }
    s *= bmask * fmask;
    if (act) s = (s > 0.f) ? s : LRELU * s;
    out[((size_t)n * CoutT + coOff + r) * HW + p] = s;
}

// ConvTranspose2d(k=4, s=2, p=1) as lhs-dilated conv; weight (2, Cin, 4, 4).
// Only 4 of 16 taps hit lattice points -> inner loop = 4 * Cin MACs, Cout = 2.
__global__ void deconv4x4_k(const float* __restrict__ x, const float* __restrict__ w,
                            const float* __restrict__ b, float* __restrict__ out,
                            int N, int Cin, int H, int W, int CoutT, int coOff)
{
    int Ho = 2 * H, Wo = 2 * W;
    int HWo = Ho * Wo;
    int i = blockIdx.x * blockDim.x + threadIdx.x;
    int tot = N * 2 * HWo;
    if (i >= tot) return;
    int p = i % HWo; int t = i / HWo; int co = t & 1; int n = t >> 1;
    int oh = p / Wo, ow = p - oh * Wo;
    float acc = b[co];
    for (int kh = 0; kh < 4; ++kh) {
        int dh = oh - 2 + kh;
        if (dh < 0 || (dh & 1)) continue;
        int ih = dh >> 1;
        if (ih >= H) continue;
        for (int kw = 0; kw < 4; ++kw) {
            int dw = ow - 2 + kw;
            if (dw < 0 || (dw & 1)) continue;
            int iw = dw >> 1;
            if (iw >= W) continue;
            const float* xr = x + ((size_t)n * Cin * H + ih) * W + iw;
            const float* wr = w + ((size_t)co * Cin * 4 + kh) * 4 + kw;
            for (int ci = 0; ci < Cin; ++ci)
                acc += xr[(size_t)ci * H * W] * wr[(size_t)ci * 16];
        }
    }
    out[((size_t)n * CoutT + coOff + co) * HWo + p] = acc;
}

__global__ void scale_k(const float* __restrict__ a, float* __restrict__ o,
                        float s, int n)
{
    int i = blockIdx.x * blockDim.x + threadIdx.x;
    if (i < n) o[i] = a[i] * s;
}

__global__ void copy_slice_k(const float* __restrict__ src, float* __restrict__ dst,
                             int N, int Cs, int H, int W, int CdT, int coOff)
{
    int HW = H * W;
    int i = blockIdx.x * blockDim.x + threadIdx.x;
    int tot = N * Cs * HW;
    if (i >= tot) return;
    int p = i % HW; int t = i / HW; int c = t % Cs; int n = t / Cs;
    dst[((size_t)n * CdT + coOff + c) * HW + p] = src[i];
}

__global__ void add_k(const float* __restrict__ a, const float* __restrict__ b,
                      float* __restrict__ o, int n)
{
    int i = blockIdx.x * blockDim.x + threadIdx.x;
    if (i < n) o[i] = a[i] + b[i];
}

// ---------------------------------------------------------------------------
// Host orchestration
// ---------------------------------------------------------------------------
struct ConvP { const float* b; const float* w; };
struct DecL  { ConvP c[5]; ConvP pf; ConvP up_flow; ConvP upfeat; };

extern "C" void kernel_launch(void* const* d_in, const int* in_sizes, int n_in,
                              void* d_out, int out_size, void* d_ws, size_t ws_size,
                              hipStream_t stream)
{
    (void)in_sizes; (void)n_in; (void)ws_size;
    // jax pytree flatten, dicts sorted by key: params(dc, dec["2".."6"], enc), x
    int idx = 0;
    auto nextP = [&](ConvP& c) {
        c.b = (const float*)d_in[idx++];
        c.w = (const float*)d_in[idx++];
    };
    ConvP dc[7];
    for (int i = 0; i < 7; ++i) nextP(dc[i]);
    DecL dec2, dec3, dec4, dec5, dec6;
    auto rdDec = [&](DecL& d, bool up) {
        for (int i = 0; i < 5; ++i) nextP(d.c[i]);
        nextP(d.pf);
        if (up) { nextP(d.up_flow); nextP(d.upfeat); }
    };
    rdDec(dec2, false); rdDec(dec3, true); rdDec(dec4, true);
    rdDec(dec5, true);  rdDec(dec6, true);
    ConvP enc[6][3];
    for (int l = 0; l < 6; ++l)
        for (int j = 0; j < 3; ++j) nextP(enc[l][j]);
    const float* x = (const float*)d_in[idx++];

    // workspace bump allocator
    char* wsp = (char*)d_ws;
    size_t wsoff = 0;
    auto alloc = [&](size_t nfloats) -> float* {
        float* p = (float*)(wsp + wsoff);
        wsoff += ((nfloats * 4 + 255) & ~(size_t)255);
        return p;
    };

    auto conv = [&](const float* in, const ConvP& p, float* o,
                    int Cin, int CinT, int ciOff, int H, int W,
                    int Cout, int CoutT, int coOff, int Ho, int Wo,
                    int stride, int pad, int dil, int act) {
        dim3 g(cdiv(2 * Ho * Wo, 32), cdiv(Cout, 64));
        conv3x3_wmma<<<g, 128, 0, stream>>>(in, p.w, p.b, o, 2, Cin, CinT, ciOff,
                                            H, W, Cout, CoutT, coOff, Ho, Wo,
                                            stride, pad, dil, act);
    };
    auto deconv = [&](const float* in, const ConvP& p, float* o,
                      int Cin, int H, int W, int CoutT, int coOff) {
        int tot = 2 * 2 * (2 * H) * (2 * W);
        deconv4x4_k<<<cdiv(tot, 256), 256, 0, stream>>>(in, p.w, p.b, o,
                                                        2, Cin, H, W, CoutT, coOff);
    };

    // ---- encoder: both images through the shared pyramid ----
    const int Hs[6] = {192, 96, 48, 24, 12, 6};
    const int Ws[6] = {320, 160, 80, 40, 20, 10};
    const int Cs[6] = {16, 32, 64, 96, 128, 196};
    float* c1f[6]; float* c2f[6];
    for (int l = 0; l < 6; ++l) {
        c1f[l] = alloc((size_t)2 * Cs[l] * Hs[l] * Ws[l]);
        c2f[l] = alloc((size_t)2 * Cs[l] * Hs[l] * Ws[l]);
    }
    float* tA = alloc(4194304);   // ping/pong temps (fits 2x128x96x160 etc.)
    float* tB = alloc(4194304);

    for (int img = 0; img < 2; ++img) {
        const float* cur = x; int curC = 3, curCT = 6, curOff = img * 3;
        int curH = 192, curW = 320;
        for (int l = 0; l < 6; ++l) {
            int s = (l == 0) ? 1 : 2;
            int Ho = Hs[l], Wo = Ws[l];
            conv(cur, enc[l][0], tA, curC, curCT, curOff, curH, curW,
                 Cs[l], Cs[l], 0, Ho, Wo, s, 1, 1, 1);
            conv(tA, enc[l][1], tB, Cs[l], Cs[l], 0, Ho, Wo,
                 Cs[l], Cs[l], 0, Ho, Wo, 1, 1, 1, 1);
            float* dst = (img == 0) ? c1f[l] : c2f[l];
            conv(tB, enc[l][2], dst, Cs[l], Cs[l], 0, Ho, Wo,
                 Cs[l], Cs[l], 0, Ho, Wo, 1, 1, 1, 1);
            cur = dst; curC = Cs[l]; curCT = Cs[l]; curOff = 0;
            curH = Ho; curW = Wo;
        }
    }

    // dense block: in-place concat (h kept as suffix channels of the buffer)
    auto dense = [&](float* buf, DecL& d, int od, int H, int W) {
        int Ctot = od + 448;
        int outs[5] = {128, 128, 96, 64, 32};
        int inOff = 448, inC = od, outOff = 448;
        for (int i = 0; i < 5; ++i) {
            outOff -= outs[i];
            conv(buf, d.c[i], buf, inC, Ctot, inOff, H, W,
                 outs[i], Ctot, outOff, H, W, 1, 1, 1, 1);
            inOff = outOff; inC += outs[i];
        }
    };

    // ---- level 6 ----
    int od6 = 169, Ctot6 = od6 + 448;
    float* dbuf6 = alloc((size_t)2 * Ctot6 * 60);
    float* f1n6 = alloc((size_t)2 * 196 * 60);
    float* f2n6 = alloc((size_t)2 * 196 * 60);
    l2norm_k<<<cdiv(2 * 60, 256), 256, 0, stream>>>(c1f[5], f1n6, 2, 196, 6, 10);
    l2norm_k<<<cdiv(2 * 60, 256), 256, 0, stream>>>(c2f[5], f2n6, 2, 196, 6, 10);
    costvol_k<<<cdiv(2 * 169 * 60, 256), 256, 0, stream>>>(
        f1n6, f2n6, (const float*)nullptr, dbuf6, 2, 196, 6, 10, 6, Ctot6, 448, 1);
    dense(dbuf6, dec6, od6, 6, 10);
    float* pf6 = alloc((size_t)2 * 2 * 60);
    conv(dbuf6, dec6.pf, pf6, Ctot6, Ctot6, 0, 6, 10, 2, 2, 0, 6, 10, 1, 1, 1, 0);
    float* flow = alloc((size_t)2 * 2 * 240);
    float* feat = alloc((size_t)2 * 2 * 240);
    deconv(pf6, dec6.up_flow, flow, 2, 6, 10, 2, 0);
    deconv(dbuf6, dec6.upfeat, feat, Ctot6, 6, 10, 2, 0);

    // ---- levels 5..2 ----
    struct Spec { DecL* d; int ci; float scale; int md; };
    Spec specs[4] = {{&dec5, 4, 0.625f, 4}, {&dec4, 3, 1.25f, 4},
                     {&dec3, 2, 2.5f, 2},  {&dec2, 1, 5.0f, 2}};
    float* h2buf = nullptr;
    for (int si = 0; si < 4; ++si) {
        DecL* d = specs[si].d;
        int ci = specs[si].ci, md = specs[si].md;
        int C = Cs[ci], H = Hs[ci], W = Ws[ci], HW = H * W;
        int K = 2 * md + 1, R = K * K;
        int od = R + 2 * C + 4, Ctot = od + 448;

        float* bm = alloc((size_t)2 * 2 * HW);
        scale_k<<<cdiv(2 * 2 * HW, 256), 256, 0, stream>>>(flow, bm,
                                                           specs[si].scale, 2 * 2 * HW);
        float* f1n = alloc((size_t)2 * C * HW);
        float* f2n = alloc((size_t)2 * C * HW);
        l2norm_k<<<cdiv(2 * HW, 256), 256, 0, stream>>>(c1f[ci], f1n, 2, C, H, W);
        l2norm_k<<<cdiv(2 * HW, 256), 256, 0, stream>>>(c2f[ci], f2n, 2, C, H, W);

        float* dbuf = alloc((size_t)2 * Ctot * HW);
        costvol_k<<<cdiv(2 * R * HW, 256), 256, 0, stream>>>(f1n, f2n, bm, dbuf,
                                                             2, C, H, W, md,
                                                             Ctot, 448, 1);
        warp_k<<<cdiv(2 * HW, 256), 256, 0, stream>>>(c1f[ci], bm, dbuf, 2, C, H, W,
                                                      -1.0f, Ctot, 448 + R);
        warp_k<<<cdiv(2 * HW, 256), 256, 0, stream>>>(c2f[ci], bm, dbuf, 2, C, H, W,
                                                      +1.0f, Ctot, 448 + R + C);
        copy_slice_k<<<cdiv(2 * 2 * HW, 256), 256, 0, stream>>>(flow, dbuf, 2, 2, H, W,
                                                                Ctot, 448 + R + 2 * C);
        copy_slice_k<<<cdiv(2 * 2 * HW, 256), 256, 0, stream>>>(feat, dbuf, 2, 2, H, W,
                                                                Ctot, 448 + R + 2 * C + 2);
        dense(dbuf, *d, od, H, W);

        if (si != 3) {
            float* pfb = alloc((size_t)2 * 2 * HW);
            conv(dbuf, d->pf, pfb, Ctot, Ctot, 0, H, W, 2, 2, 0, H, W, 1, 1, 1, 0);
            float* nflow = alloc((size_t)2 * 2 * 4 * HW);
            float* nfeat = alloc((size_t)2 * 2 * 4 * HW);
            deconv(pfb, d->up_flow, nflow, 2, H, W, 2, 0);
            deconv(dbuf, d->upfeat, nfeat, Ctot, H, W, 2, 0);
            flow = nflow; feat = nfeat;
        } else {
            h2buf = dbuf;
        }
    }

    // ---- final: flow2 + dilated-conv refinement chain ----
    int H2 = 96, W2 = 160, HW2 = H2 * W2;
    float* flow2 = alloc((size_t)2 * 2 * HW2);
    conv(h2buf, dec2.pf, flow2, 541, 541, 0, H2, W2, 2, 2, 0, H2, W2, 1, 1, 1, 0);

    const int dils[6]  = {1, 2, 4, 8, 16, 1};
    const int cins[7]  = {541, 128, 128, 128, 96, 64, 32};
    const int couts[7] = {128, 128, 128, 96, 64, 32, 2};
    const float* g = h2buf;
    float* gt[2] = {tA, tB};
    for (int i = 0; i < 6; ++i) {
        float* o = gt[i & 1];
        conv(g, dc[i], o, cins[i], cins[i], 0, H2, W2,
             couts[i], couts[i], 0, H2, W2, 1, dils[i], dils[i], 1);
        g = o;
    }
    float* gout = alloc((size_t)2 * 2 * HW2);
    conv(g, dc[6], gout, 32, 32, 0, H2, W2, 2, 2, 0, H2, W2, 1, 1, 1, 0);

    add_k<<<cdiv(out_size, 256), 256, 0, stream>>>(flow2, gout, (float*)d_out, out_size);
}
